// EncoderLayer_12266426597997
// MI455X (gfx1250) — matrix operand order repaired and checked by reading, not compiled, not run on hardware
//
#include <hip/hip_runtime.h>
#include <cstdint>

typedef _Float16 v16h __attribute__((ext_vector_type(16)));
typedef float    v8f  __attribute__((ext_vector_type(8)));

#define D_MODEL 1024
#define D_FF    4096
#define NHEAD   16
#define DKDIM   64
#define BATCH   16
#define SEQ     512
#define MTOT    (BATCH*SEQ)      /* 8192 */

#define TILE_M 128
#define TILE_N 128
#define TILE_K 64
#define LDSP   72                /* padded K-stride in halves (144B rows, 16B aligned) */

// ---------------------------------------------------------------------------
// f32 -> f16 conversion (grid-stride)
// ---------------------------------------------------------------------------
__global__ __launch_bounds__(256) void cvt_f32_f16(const float* __restrict__ s,
                                                   _Float16* __restrict__ d,
                                                   long long n) {
    long long i = (long long)blockIdx.x * blockDim.x + threadIdx.x;
    long long stride = (long long)gridDim.x * blockDim.x;
    for (; i < n; i += stride) d[i] = (_Float16)s[i];
}

// ---------------------------------------------------------------------------
// Tiled transpose to f16: out[c*ldout + r] = (f16) in[r*ldin + c]
// Grid: (C/32, R/32, Z) with (batch,head) = (z/Hpb, z%Hpb) offsets.
// Dims must be multiples of 32 (true for all uses: 1024/4096/512/64).
// ---------------------------------------------------------------------------
template<typename TIN>
__global__ __launch_bounds__(256) void transpose_to_f16(
    const TIN* __restrict__ in, _Float16* __restrict__ outp,
    long long ldin, long long ldout,
    long long offin_b, long long offin_h,
    long long offout_b, long long offout_h, int Hpb)
{
    __shared__ _Float16 tile[32][33];
    const int z  = blockIdx.z;
    const int bb = z / Hpb;
    const int hh = z % Hpb;
    const TIN* ip = in + (long long)bb * offin_b + (long long)hh * offin_h;
    _Float16*  op = outp + (long long)bb * offout_b + (long long)hh * offout_h;
    const int c0 = blockIdx.x * 32;
    const int r0 = blockIdx.y * 32;
    const int tx = threadIdx.x & 31;
    const int ty = threadIdx.x >> 5;   // 0..7
#pragma unroll
    for (int i = 0; i < 4; ++i) {
        const int r = r0 + ty + i * 8;
        tile[ty + i * 8][tx] = (_Float16)ip[(long long)r * ldin + (c0 + tx)];
    }
    __syncthreads();
#pragma unroll
    for (int i = 0; i < 4; ++i) {
        const int c = c0 + ty + i * 8;     // output row = original col
        op[(long long)c * ldout + (r0 + tx)] = tile[tx][ty + i * 8];
    }
}

// ---------------------------------------------------------------------------
// WMMA GEMM: C[M,N] = A[M,K] * B^T  with B stored row-major as [N][K]
//   AF32 : A stored as f32 (converted while staging to LDS)
//   Batched over blockIdx.z with (batch,head) = (z / Hpb, z % Hpb) offsets.
// Block: 256 threads (8 waves). Wave (wm,wn) computes a 32x64 tile as 2x4
// 16x16 WMMA accumulators. K-tile = 64 (two 16x16x32 substeps). Direct
// global->LDS staging (transient regs only; an 8-wave block is capped near
// 128 VGPRs/wave, so persistent staging buffers would spill to scratch).
// Next K-tile is prefetched into cache (global_prefetch_b8) during compute.
// Assumes M % 128 == 0, K % 64 == 0, N % 16 == 0 (true for every launch).
// ---------------------------------------------------------------------------
template<bool AF32, bool BIAS, bool RELU, bool MASKSC, bool OUTH>
__global__ __launch_bounds__(256, 1) void gemm_wmma(
    const void* __restrict__ Agv, const _Float16* __restrict__ Bg0,
    const float* __restrict__ bias,
    float* __restrict__ Cf0, _Float16* __restrict__ Ch0,
    int M, int N, int K,
    long long lda, long long ldb, long long ldc,
    long long offA_b, long long offA_h,
    long long offB_b, long long offB_h,
    long long offC_b, long long offC_h,
    int Hpb, const int* __restrict__ maskp, int maskLen, float scale)
{
    __shared__ _Float16 Alds[TILE_M * LDSP];
    __shared__ _Float16 Blds[TILE_N * LDSP];

    const int t    = threadIdx.x;
    const int lane = t & 31;
    const int wave = t >> 5;
    const int wm   = wave >> 1;   // 0..3 -> 32-row slab
    const int wn   = wave & 1;    // 0..1 -> 64-col slab
    const int grp  = lane >> 4;   // 0/1
    const int l16  = lane & 15;

    const int n0 = blockIdx.x * TILE_N;
    const int m0 = blockIdx.y * TILE_M;
    const int z  = blockIdx.z;
    const int bb = z / Hpb;
    const int hh = z % Hpb;

    const _Float16* Ah = nullptr;
    const float*    Af = nullptr;
    if constexpr (AF32) Af = (const float*)Agv + (long long)bb * offA_b + (long long)hh * offA_h;
    else                Ah = (const _Float16*)Agv + (long long)bb * offA_b + (long long)hh * offA_h;
    const _Float16* Bg = Bg0 + (long long)bb * offB_b + (long long)hh * offB_h;

    // Both A and B tiles use the same [row][K-contiguous] staging pattern:
    // 2 threads per row, 32 halves (64B) each.
    const int ar = t >> 1;            // A row / B column-row
    const int ac = (t & 1) << 5;      // 0 or 32 halves along K
    const bool bInRange = (n0 + ar) < N;

    auto stageA = [&](int k0) {
        if constexpr (AF32) {
            const float4* src = (const float4*)(Af + (long long)(m0 + ar) * lda + (k0 + ac));
            float4 f0 = src[0], f1 = src[1], f2 = src[2], f3 = src[3];
            float4 f4 = src[4], f5 = src[5], f6 = src[6], f7 = src[7];
            union { _Float16 h[32]; uint4 q[4]; } tv;
            tv.h[0]=(_Float16)f0.x; tv.h[1]=(_Float16)f0.y; tv.h[2]=(_Float16)f0.z; tv.h[3]=(_Float16)f0.w;
            tv.h[4]=(_Float16)f1.x; tv.h[5]=(_Float16)f1.y; tv.h[6]=(_Float16)f1.z; tv.h[7]=(_Float16)f1.w;
            tv.h[8]=(_Float16)f2.x; tv.h[9]=(_Float16)f2.y; tv.h[10]=(_Float16)f2.z; tv.h[11]=(_Float16)f2.w;
            tv.h[12]=(_Float16)f3.x; tv.h[13]=(_Float16)f3.y; tv.h[14]=(_Float16)f3.z; tv.h[15]=(_Float16)f3.w;
            tv.h[16]=(_Float16)f4.x; tv.h[17]=(_Float16)f4.y; tv.h[18]=(_Float16)f4.z; tv.h[19]=(_Float16)f4.w;
            tv.h[20]=(_Float16)f5.x; tv.h[21]=(_Float16)f5.y; tv.h[22]=(_Float16)f5.z; tv.h[23]=(_Float16)f5.w;
            tv.h[24]=(_Float16)f6.x; tv.h[25]=(_Float16)f6.y; tv.h[26]=(_Float16)f6.z; tv.h[27]=(_Float16)f6.w;
            tv.h[28]=(_Float16)f7.x; tv.h[29]=(_Float16)f7.y; tv.h[30]=(_Float16)f7.z; tv.h[31]=(_Float16)f7.w;
#pragma unroll
            for (int i = 0; i < 4; ++i) *(uint4*)&Alds[ar * LDSP + ac + 8*i] = tv.q[i];
        } else {
            const uint4* src = (const uint4*)(Ah + (long long)(m0 + ar) * lda + (k0 + ac));
            uint4 v0 = src[0], v1 = src[1], v2 = src[2], v3 = src[3];
            *(uint4*)&Alds[ar * LDSP + ac]      = v0;
            *(uint4*)&Alds[ar * LDSP + ac + 8]  = v1;
            *(uint4*)&Alds[ar * LDSP + ac + 16] = v2;
            *(uint4*)&Alds[ar * LDSP + ac + 24] = v3;
        }
    };
    auto stageB = [&](int k0) {
        uint4 v0 = (uint4){0,0,0,0}, v1 = v0, v2 = v0, v3 = v0;
        if (bInRange) {
            const uint4* src = (const uint4*)(Bg + (long long)(n0 + ar) * ldb + (k0 + ac));
            v0 = src[0]; v1 = src[1]; v2 = src[2]; v3 = src[3];
        }
        *(uint4*)&Blds[ar * LDSP + ac]      = v0;
        *(uint4*)&Blds[ar * LDSP + ac + 8]  = v1;
        *(uint4*)&Blds[ar * LDSP + ac + 16] = v2;
        *(uint4*)&Blds[ar * LDSP + ac + 24] = v3;
    };

    v8f acc[2][4];
#pragma unroll
    for (int mi = 0; mi < 2; ++mi)
#pragma unroll
        for (int ni = 0; ni < 4; ++ni) acc[mi][ni] = (v8f){};

    for (int k0 = 0; k0 < K; k0 += TILE_K) {
        stageA(k0);
        stageB(k0);
        // prefetch the next K-tile into cache while this tile is computed
        if (k0 + TILE_K < K) {
            if constexpr (AF32)
                __builtin_prefetch(Af + (long long)(m0 + ar) * lda + (k0 + TILE_K + ac), 0, 1);
            else
                __builtin_prefetch(Ah + (long long)(m0 + ar) * lda + (k0 + TILE_K + ac), 0, 1);
            if (bInRange)
                __builtin_prefetch(Bg + (long long)(n0 + ar) * ldb + (k0 + TILE_K + ac), 0, 1);
        }
        __syncthreads();

        // two 16x16x32 K-substeps per LDS tile
#pragma unroll
        for (int ks = 0; ks < TILE_K; ks += 32) {
            v16h afr[2];
            v16h bfr[4];
#pragma unroll
            for (int mi = 0; mi < 2; ++mi) {
                const _Float16* ap = &Alds[(wm * 32 + mi * 16 + l16) * LDSP + ks];
                union { v16h v; unsigned int u[8]; } au;
#pragma unroll
                for (int p = 0; p < 4; ++p) {
                    au.u[p]     = *(const unsigned int*)&ap[2 * p + grp * 8];
                    au.u[4 + p] = *(const unsigned int*)&ap[16 + 2 * p + grp * 8];
                }
                afr[mi] = au.v;
            }
#pragma unroll
            for (int ni = 0; ni < 4; ++ni) {
                const _Float16* bp = &Blds[(wn * 64 + ni * 16 + l16) * LDSP + ks + grp * 8];
                union { v16h v; uint4 q[2]; } bu;
                bu.q[0] = *(const uint4*)&bp[0];
                bu.q[1] = *(const uint4*)&bp[16];
                bfr[ni] = bu.v;
            }
#pragma unroll
            for (int ni = 0; ni < 4; ++ni) {
                acc[0][ni] = __builtin_amdgcn_wmma_f32_16x16x32_f16(
                    false, afr[0], false, bfr[ni], (short)0, acc[0][ni], false, false);
                acc[1][ni] = __builtin_amdgcn_wmma_f32_16x16x32_f16(
                    false, afr[1], false, bfr[ni], (short)0, acc[1][ni], false, false);
            }
        }
        __syncthreads();
    }

    // ---- epilogue
    float*    Cf = nullptr;
    _Float16* Ch = nullptr;
    if constexpr (OUTH) Ch = Ch0 + (long long)bb * offC_b + (long long)hh * offC_h;
    else                Cf = Cf0 + (long long)bb * offC_b + (long long)hh * offC_h;
    const int* mrow = nullptr;
    if constexpr (MASKSC) mrow = maskp + (long long)bb * maskLen;

#pragma unroll
    for (int mi = 0; mi < 2; ++mi) {
#pragma unroll
        for (int ni = 0; ni < 4; ++ni) {
            const int col = n0 + wn * 64 + ni * 16 + l16;
            if (col < N) {
                float bv = 0.0f;
                if constexpr (BIAS) bv = bias[col];
                bool masked = false;
                if constexpr (MASKSC) masked = (mrow[col] == 0);
#pragma unroll
                for (int r = 0; r < 8; ++r) {
                    const int row = m0 + wm * 32 + mi * 16 + grp * 8 + r;
                    float v = acc[mi][ni][r];
                    if constexpr (BIAS) v += bv;
                    if constexpr (MASKSC) { v *= scale; if (masked) v = -1e9f; }
                    if constexpr (RELU) v = fmaxf(v, 0.0f);
                    const long long off = (long long)row * ldc + col;
                    if constexpr (OUTH) Ch[off] = (_Float16)v;
                    else                Cf[off] = v;
                }
            }
        }
    }
}

// ---------------------------------------------------------------------------
// In-place row softmax over 512 columns (one block per row)
// ---------------------------------------------------------------------------
__global__ __launch_bounds__(256) void softmax512(float* __restrict__ attn) {
    float* p = attn + (long long)blockIdx.x * 512;
    const int t = threadIdx.x;
    __shared__ float red[256];
    float v0 = p[t], v1 = p[t + 256];
    red[t] = fmaxf(v0, v1);
    __syncthreads();
    for (int s = 128; s > 0; s >>= 1) {
        if (t < s) red[t] = fmaxf(red[t], red[t + s]);
        __syncthreads();
    }
    const float m = red[0];
    __syncthreads();
    const float e0 = __expf(v0 - m), e1 = __expf(v1 - m);
    red[t] = e0 + e1;
    __syncthreads();
    for (int s = 128; s > 0; s >>= 1) {
        if (t < s) red[t] += red[t + s];
        __syncthreads();
    }
    const float inv = 1.0f / red[0];
    p[t] = e0 * inv;
    p[t + 256] = e1 * inv;
}

// ---------------------------------------------------------------------------
// LayerNorm over 1024 cols with fused residual add; optional f16 mirror out
// ---------------------------------------------------------------------------
__global__ __launch_bounds__(256) void layernorm1024(
    const float* __restrict__ inp, const float* __restrict__ res,
    const float* __restrict__ g, const float* __restrict__ bta,
    float* __restrict__ out32, _Float16* __restrict__ out16)
{
    const long long row = blockIdx.x;
    const int t = threadIdx.x;
    const float* a = inp + row * D_MODEL;
    const float* r = res + row * D_MODEL;
    float x[4];
    float s = 0.0f, q = 0.0f;
#pragma unroll
    for (int i = 0; i < 4; ++i) {
        x[i] = a[t + 256 * i] + r[t + 256 * i];
        s += x[i];
        q += x[i] * x[i];
    }
    __shared__ float rs[256];
    __shared__ float rq[256];
    rs[t] = s; rq[t] = q;
    __syncthreads();
    for (int st = 128; st > 0; st >>= 1) {
        if (t < st) { rs[t] += rs[t + st]; rq[t] += rq[t + st]; }
        __syncthreads();
    }
    const float mean = rs[0] * (1.0f / D_MODEL);
    const float var  = rq[0] * (1.0f / D_MODEL) - mean * mean;
    const float rstd = rsqrtf(var + 1e-6f);
#pragma unroll
    for (int i = 0; i < 4; ++i) {
        const int c = t + 256 * i;
        const float y = (x[i] - mean) * rstd * g[c] + bta[c];
        out32[row * D_MODEL + c] = y;
        if (out16) out16[row * D_MODEL + c] = (_Float16)y;
    }
}

// ---------------------------------------------------------------------------
// Host launcher
// ---------------------------------------------------------------------------
extern "C" void kernel_launch(void* const* d_in, const int* in_sizes, int n_in,
                              void* d_out, int out_size, void* d_ws, size_t ws_size,
                              hipStream_t stream) {
    (void)in_sizes; (void)n_in; (void)out_size; (void)ws_size;

    const float* X    = (const float*)d_in[0];
    const float* Wq   = (const float*)d_in[1];
    const float* bq   = (const float*)d_in[2];
    const float* Wk   = (const float*)d_in[3];
    const float* bk   = (const float*)d_in[4];
    const float* Wv   = (const float*)d_in[5];
    const float* bv   = (const float*)d_in[6];
    const float* Wo   = (const float*)d_in[7];
    const float* bo   = (const float*)d_in[8];
    const float* ln1g = (const float*)d_in[9];
    const float* ln1b = (const float*)d_in[10];
    const float* w1   = (const float*)d_in[11];
    const float* b1   = (const float*)d_in[12];
    const float* w2   = (const float*)d_in[13];
    const float* b2   = (const float*)d_in[14];
    const float* ln2g = (const float*)d_in[15];
    const float* ln2b = (const float*)d_in[16];
    const int*   mask = (const int*)d_in[17];

    float* out  = (float*)d_out;
    float* attn = out + (long long)MTOT * D_MODEL;  // [B,H,S,S]

    // workspace carve-up (256B aligned)
    char* w = (char*)d_ws;
    size_t off = 0;
    auto alloc = [&](size_t bytes) -> void* {
        off = (off + 255) & ~(size_t)255;
        void* p = w + off;
        off += bytes;
        return p;
    };
    _Float16* Xh   = (_Float16*)alloc((size_t)MTOT * D_MODEL * 2);
    _Float16* WqT  = (_Float16*)alloc((size_t)D_MODEL * D_MODEL * 2);  // [N][K]
    _Float16* WkT  = (_Float16*)alloc((size_t)D_MODEL * D_MODEL * 2);
    _Float16* WvT  = (_Float16*)alloc((size_t)D_MODEL * D_MODEL * 2);
    _Float16* WoT  = (_Float16*)alloc((size_t)D_MODEL * D_MODEL * 2);
    _Float16* w1T  = (_Float16*)alloc((size_t)D_MODEL * D_FF * 2);     // [4096][1024]
    _Float16* w2T  = (_Float16*)alloc((size_t)D_FF * D_MODEL * 2);     // [1024][4096]
    _Float16* Qh   = (_Float16*)alloc((size_t)MTOT * D_MODEL * 2);
    _Float16* Kh   = (_Float16*)alloc((size_t)MTOT * D_MODEL * 2);
    _Float16* Vh   = (_Float16*)alloc((size_t)MTOT * D_MODEL * 2);
    _Float16* Vt   = (_Float16*)alloc((size_t)MTOT * D_MODEL * 2);     // [B,H,64,512]
    _Float16* ctxh = (_Float16*)alloc((size_t)MTOT * D_MODEL * 2);
    float*    y1   = (float*)alloc((size_t)MTOT * D_MODEL * 4);
    float*    x2   = (float*)alloc((size_t)MTOT * D_MODEL * 4);
    _Float16* x2h  = (_Float16*)alloc((size_t)MTOT * D_MODEL * 2);
    _Float16* ffh  = (_Float16*)alloc((size_t)MTOT * D_FF * 2);
    float*    ff2  = (float*)alloc((size_t)MTOT * D_MODEL * 4);

    const dim3 cb(256);

    // ---- X f32 -> f16 (row-major, A operand)
    cvt_f32_f16<<<dim3(4096), cb, 0, stream>>>(X, Xh, (long long)MTOT * D_MODEL);

    // ---- weights: f32 -> f16 with transpose to [N][K]
    transpose_to_f16<float><<<dim3(32, 32, 1), cb, 0, stream>>>(
        Wq, WqT, D_MODEL, D_MODEL, 0, 0, 0, 0, 1);
    transpose_to_f16<float><<<dim3(32, 32, 1), cb, 0, stream>>>(
        Wk, WkT, D_MODEL, D_MODEL, 0, 0, 0, 0, 1);
    transpose_to_f16<float><<<dim3(32, 32, 1), cb, 0, stream>>>(
        Wv, WvT, D_MODEL, D_MODEL, 0, 0, 0, 0, 1);
    transpose_to_f16<float><<<dim3(32, 32, 1), cb, 0, stream>>>(
        Wo, WoT, D_MODEL, D_MODEL, 0, 0, 0, 0, 1);
    transpose_to_f16<float><<<dim3(D_FF / 32, 32, 1), cb, 0, stream>>>(
        w1, w1T, D_FF, D_MODEL, 0, 0, 0, 0, 1);
    transpose_to_f16<float><<<dim3(32, D_FF / 32, 1), cb, 0, stream>>>(
        w2, w2T, D_MODEL, D_FF, 0, 0, 0, 0, 1);

    // ---- Q/K/V projections: [8192,1024] = Xh @ W^T(stored [N][K]) + b -> f16
    {
        dim3 grid(D_MODEL / TILE_N, MTOT / TILE_M, 1);
        gemm_wmma<false, true, false, false, true><<<grid, cb, 0, stream>>>(
            Xh, WqT, bq, nullptr, Qh, MTOT, D_MODEL, D_MODEL,
            D_MODEL, D_MODEL, D_MODEL, 0, 0, 0, 0, 0, 0, 1, nullptr, 0, 0.0f);
        gemm_wmma<false, true, false, false, true><<<grid, cb, 0, stream>>>(
            Xh, WkT, bk, nullptr, Kh, MTOT, D_MODEL, D_MODEL,
            D_MODEL, D_MODEL, D_MODEL, 0, 0, 0, 0, 0, 0, 1, nullptr, 0, 0.0f);
        gemm_wmma<false, true, false, false, true><<<grid, cb, 0, stream>>>(
            Xh, WvT, bv, nullptr, Vh, MTOT, D_MODEL, D_MODEL,
            D_MODEL, D_MODEL, D_MODEL, 0, 0, 0, 0, 0, 0, 1, nullptr, 0, 0.0f);
    }

    // ---- V per-head transpose: Vh[B,S,H*64] -> Vt[B,H,64,512]
    transpose_to_f16<_Float16><<<dim3(DKDIM / 32, SEQ / 32, BATCH * NHEAD), cb, 0, stream>>>(
        Vh, Vt, D_MODEL, SEQ,
        (long long)SEQ * D_MODEL, DKDIM,
        (long long)NHEAD * DKDIM * SEQ, (long long)DKDIM * SEQ, NHEAD);

    // ---- scores = Q @ K^T / sqrt(dk), masked  -> attn region (f32)
    {
        dim3 grid(SEQ / TILE_N, SEQ / TILE_M, BATCH * NHEAD);
        gemm_wmma<false, false, false, true, false><<<grid, cb, 0, stream>>>(
            Qh, Kh, nullptr, attn, nullptr, SEQ, SEQ, DKDIM,
            D_MODEL, D_MODEL, SEQ,
            (long long)SEQ * D_MODEL, DKDIM,
            (long long)SEQ * D_MODEL, DKDIM,
            (long long)NHEAD * SEQ * SEQ, (long long)SEQ * SEQ,
            NHEAD, mask, SEQ, 0.125f);
    }

    // ---- softmax rows (B*H*S rows of 512)
    softmax512<<<dim3(BATCH * NHEAD * SEQ), cb, 0, stream>>>(attn);

    // ---- ctx = attn @ V   (A is f32 probs, B = Vt [64][512]) -> ctxh f16
    {
        dim3 grid(1, SEQ / TILE_M, BATCH * NHEAD);
        gemm_wmma<true, false, false, false, true><<<grid, cb, 0, stream>>>(
            attn, Vt, nullptr, nullptr, ctxh, SEQ, DKDIM, SEQ,
            SEQ, SEQ, D_MODEL,
            (long long)NHEAD * SEQ * SEQ, (long long)SEQ * SEQ,
            (long long)NHEAD * DKDIM * SEQ, (long long)DKDIM * SEQ,
            (long long)SEQ * D_MODEL, DKDIM,
            NHEAD, nullptr, 0, 0.0f);
    }

    // ---- y1 = ctx @ Wo + bo -> f32
    {
        dim3 grid(D_MODEL / TILE_N, MTOT / TILE_M, 1);
        gemm_wmma<false, true, false, false, false><<<grid, cb, 0, stream>>>(
            ctxh, WoT, bo, y1, nullptr, MTOT, D_MODEL, D_MODEL,
            D_MODEL, D_MODEL, D_MODEL, 0, 0, 0, 0, 0, 0, 1, nullptr, 0, 0.0f);
    }

    // ---- x2 = LN1(y1 + X)  (f32 + f16 mirror)
    layernorm1024<<<dim3(MTOT), cb, 0, stream>>>(y1, X, ln1g, ln1b, x2, x2h);

    // ---- ff = relu(x2 @ w1 + b1) -> f16 [8192,4096]
    {
        dim3 grid(D_FF / TILE_N, MTOT / TILE_M, 1);
        gemm_wmma<false, true, true, false, true><<<grid, cb, 0, stream>>>(
            x2h, w1T, b1, nullptr, ffh, MTOT, D_FF, D_MODEL,
            D_MODEL, D_MODEL, D_FF, 0, 0, 0, 0, 0, 0, 1, nullptr, 0, 0.0f);
    }

    // ---- ff2 = ff @ w2 + b2 -> f32
    {
        dim3 grid(D_MODEL / TILE_N, MTOT / TILE_M, 1);
        gemm_wmma<false, true, false, false, false><<<grid, cb, 0, stream>>>(
            ffh, w2T, b2, ff2, nullptr, MTOT, D_MODEL, D_FF,
            D_FF, D_FF, D_MODEL, 0, 0, 0, 0, 0, 0, 1, nullptr, 0, 0.0f);
    }

    // ---- out = LN2(ff2 + x2) -> d_out (f32)
    layernorm1024<<<dim3(MTOT), cb, 0, stream>>>(ff2, x2, ln2g, ln2b, out, nullptr);
}
